// SelfAttention_12635793785267
// MI455X (gfx1250) — compile-verified, matrix operand order checked by reading, not run on hardware
//
#include <hip/hip_runtime.h>
#include <hip/hip_bf16.h>
#include <math.h>
#include <stdint.h>

// ---------------------------------------------------------------------------
// MI455X (gfx1250) implementation. wave32, WMMA f16 16x16x32 with f32 accum
// for all GEMM-shaped work; TDM (tensor_load_to_lds) + TENSORcnt double
// buffering feeds the big GEMMs; radix-2 LDS FFT for 16384-point row FFTs.
// ---------------------------------------------------------------------------

typedef __attribute__((ext_vector_type(16))) _Float16 v16h;
typedef __attribute__((ext_vector_type(8)))  _Float16 v8h;
typedef __attribute__((ext_vector_type(8)))  float    v8f;

#define HW   16384   // H*W = 128*128
#define NC   256     // channels
#define NB   4       // batch

#if defined(__has_builtin)
#if __has_builtin(__builtin_amdgcn_tensor_load_to_lds) && \
    __has_builtin(__builtin_amdgcn_s_wait_tensorcnt)
#define USE_TDM 1
#endif
#endif
#ifndef USE_TDM
#define USE_TDM 0
#endif

// K-index inside a 16x32 f16 WMMA operand for element j (0..15) of a lane,
// per the CDNA5 ISA 16-bit A/B layout (lanes 0-15: K {0..7,16..23},
// lanes 16-31: K {8..15,24..31}).
__device__ __forceinline__ int wmma_kidx(int j, int half) {
  return j + 8 * half + ((j >= 8) ? 8 : 0);
}

#if USE_TDM
typedef unsigned int v4u __attribute__((ext_vector_type(4)));
typedef int          v4i __attribute__((ext_vector_type(4)));
typedef int          v8i __attribute__((ext_vector_type(8)));

// Issue one TDM 2D f16 tile load: tileK rows (dim1) x tileN elems (dim0,
// contiguous), row stride strideElems, into LDS at byte offset ldsByteOff.
// Descriptor per CDNA5 ISA 8.3 (group0) / 8.4 (group1); groups 2/3 zero (2D).
// 6-arg builtin form (clang-23 / therock headers): the 5th arg is an extra
// zero-filled int32x8 group; cpol immediate last.
__device__ __forceinline__ void tdm_load_f16_tile(const _Float16* gptr,
                                                  unsigned ldsByteOff,
                                                  unsigned tileN, unsigned tileK,
                                                  unsigned strideElems,
                                                  unsigned tensorN, unsigned tensorK) {
  unsigned long long ga = (unsigned long long)(uintptr_t)gptr;
  v4u g0;
  g0[0] = 1u;                                        // count=1 (valid user D#)
  g0[1] = ldsByteOff;                                // lds_addr (bytes)
  g0[2] = (unsigned)(ga & 0xffffffffu);              // global_addr[31:0]
  g0[3] = (unsigned)((ga >> 32) & 0x01ffffffu)       // global_addr[56:32]
          | (2u << 30);                              // type=2 ("image")
  v8i g1;
  g1[0] = (int)(1u << 16);                           // data_size=1 -> 2 bytes
  g1[1] = (int)((tensorN & 0xffffu) << 16);          // tensor_dim0[15:0]
  g1[2] = (int)((tensorN >> 16) & 0xffffu)           // tensor_dim0[31:16]
          | (int)((tensorK & 0xffffu) << 16);        // tensor_dim1[15:0]
  g1[3] = (int)((tensorK >> 16) & 0xffffu)           // tensor_dim1[31:16]
          | (int)(tileN << 16);                      // tile_dim0
  g1[4] = (int)tileK;                                // tile_dim1 (tile_dim2=0)
  g1[5] = (int)strideElems;                          // tensor_dim0_stride[31:0]
  g1[6] = 0;                                         // stride0 hi / stride1 lo
  g1[7] = 0;
  v4i z4 = {0, 0, 0, 0};
  v8i z8 = {0, 0, 0, 0, 0, 0, 0, 0};
  __builtin_amdgcn_tensor_load_to_lds(g0, g1, z4, z4, z8, 0);
}
#endif

// ---------------------------------------------------------------------------
// 0) One-time f32 -> f16 weight conversion
// ---------------------------------------------------------------------------
__global__ void k_cvt_f32_to_f16(const float* __restrict__ src,
                                 _Float16* __restrict__ dst, int n) {
  int i = blockIdx.x * 256 + threadIdx.x;
  if (i < n) dst[i] = (_Float16)src[i];
}

// ---------------------------------------------------------------------------
// 1) GroupNorm (32 groups of 8 channels) -> f16 for the WMMA GEMM
// ---------------------------------------------------------------------------
__global__ void k_groupnorm(const float* __restrict__ x, const float* __restrict__ gw,
                            const float* __restrict__ gb, _Float16* __restrict__ xn) {
  const int b = blockIdx.x >> 5;
  const int g = blockIdx.x & 31;
  const size_t base = ((size_t)b * NC + g * 8) * HW;
  const int n = 8 * HW;
  float s = 0.f, s2 = 0.f;
  for (int i = threadIdx.x; i < n; i += 256) {
    float v = x[base + i];
    s += v; s2 += v * v;
  }
  __shared__ float r0[256], r1[256];
  r0[threadIdx.x] = s; r1[threadIdx.x] = s2;
  __syncthreads();
  for (int off = 128; off > 0; off >>= 1) {
    if (threadIdx.x < off) { r0[threadIdx.x] += r0[threadIdx.x + off];
                             r1[threadIdx.x] += r1[threadIdx.x + off]; }
    __syncthreads();
  }
  const float mean = r0[0] / n;
  const float var  = r1[0] / n - mean * mean;
  const float rstd = rsqrtf(var + 1e-5f);
  for (int i = threadIdx.x; i < n; i += 256) {
    int ch = g * 8 + (i >> 14);
    float v = (x[base + i] - mean) * rstd * gw[ch] + gb[ch];
    xn[base + i] = (_Float16)v;
  }
}

// ---------------------------------------------------------------------------
// GEMM core used by the two 1x1 convs: D[m0..+16, n0..+16] over K=256.
// A (weights, f16): two contiguous 16B loads per lane per step.
// B (activations): TDM tile -> per-wave LDS slice, 2-deep pipeline.
// ---------------------------------------------------------------------------
__device__ __forceinline__ v8f gemm_k256_tile(const _Float16* __restrict__ Wh,
                                              const _Float16* __restrict__ X,
                                              _Float16* __restrict__ btile0,
                                              _Float16* __restrict__ btile1,
                                              int m0, int n0, int half, int r) {
  const _Float16* Arow = Wh + (m0 + r) * NC;
  v8f acc = {};
#if USE_TDM
  const unsigned lds0 = (unsigned)(uintptr_t)btile0;
  const unsigned lds1 = (unsigned)(uintptr_t)btile1;
  tdm_load_f16_tile(X + n0, lds0, 16, 32, HW, HW, NC);   // k0=0 tile
#endif
  for (int k0 = 0, it = 0; k0 < NC; k0 += 32, ++it) {
    v16h av, bv;
    v8h a0 = *(const v8h*)(Arow + k0 + 8 * half);
    v8h a1 = *(const v8h*)(Arow + k0 + 16 + 8 * half);
#pragma unroll
    for (int j = 0; j < 8; ++j) { av[j] = a0[j]; av[j + 8] = a1[j]; }
#if USE_TDM
    if (k0 + 32 < NC) {           // prefetch next K-tile into the other slice
      tdm_load_f16_tile(X + (size_t)(k0 + 32) * HW + n0,
                        (it & 1) ? lds0 : lds1, 16, 32, HW, HW, NC);
      __builtin_amdgcn_s_wait_tensorcnt(1);
    } else {
      __builtin_amdgcn_s_wait_tensorcnt(0);
    }
    const _Float16* bt = (it & 1) ? btile1 : btile0;   // [K=32][N=16] row-major
#pragma unroll
    for (int j = 0; j < 16; ++j)
      bv[j] = bt[wmma_kidx(j, half) * 16 + r];
#else
#pragma unroll
    for (int j = 0; j < 16; ++j)
      bv[j] = X[(size_t)(k0 + wmma_kidx(j, half)) * HW + n0 + r];
#endif
    acc = __builtin_amdgcn_wmma_f32_16x16x32_f16(false, av, false, bv,
                                                 (short)0, acc, false, false);
  }
  return acc;
}

// ---------------------------------------------------------------------------
// 2) QKV 1x1 conv: qkv[b,o,s] = sum_i Wq[o,i]*xn[b,i,s]  (M=768,N=16384,K=256)
// ---------------------------------------------------------------------------
__global__ void k_qkv_gemm(const _Float16* __restrict__ Wh, const _Float16* __restrict__ xn,
                           float* __restrict__ qkv) {
  __shared__ _Float16 btile[8][2][32 * 16];
  const int wave = threadIdx.x >> 5, lane = threadIdx.x & 31;
  const int half = lane >> 4, r = lane & 15;
  const int n0 = blockIdx.x * 128 + wave * 16;
  const int m0 = blockIdx.y * 16;
  const int b  = blockIdx.z;
  const _Float16* X = xn + (size_t)b * NC * HW;
  v8f acc = gemm_k256_tile(Wh, X, &btile[wave][0][0], &btile[wave][1][0],
                           m0, n0, half, r);
  float* O = qkv + (size_t)b * 768 * HW;
#pragma unroll
  for (int t = 0; t < 8; ++t)
    O[(size_t)(m0 + t + 8 * half) * HW + n0 + r] = acc[t];
}

// ---------------------------------------------------------------------------
// 3) 3x3 depthwise conv, pad 1, groups=768
// ---------------------------------------------------------------------------
__global__ void k_dwconv(const float* __restrict__ in, const float* __restrict__ w9,
                         float* __restrict__ out) {
  const int hw = blockIdx.x * 256 + threadIdx.x;
  const int ch = blockIdx.y, b = blockIdx.z;
  const int y = hw >> 7, xx = hw & 127;
  const float* wp = w9 + ch * 9;
  const size_t base = ((size_t)b * 768 + ch) * HW;
  float s = 0.f;
#pragma unroll
  for (int dy = -1; dy <= 1; ++dy)
#pragma unroll
    for (int dx = -1; dx <= 1; ++dx) {
      int yy = y + dy, xc = xx + dx;
      if (yy >= 0 && yy < 128 && xc >= 0 && xc < 128)
        s += in[base + yy * 128 + xc] * wp[(dy + 1) * 3 + dx + 1];
    }
  out[base + hw] = s;
}

// ---------------------------------------------------------------------------
// 4) L2-normalize each q,k row (channels 0..511) over HW, in place
// ---------------------------------------------------------------------------
__global__ void k_l2norm(float* __restrict__ qk) {
  const int row = blockIdx.x;            // 0..2047
  const int b = row >> 9, c = row & 511; // c<256:q, 256..511:k
  float* p = qk + ((size_t)b * 768 + c) * HW;
  float s = 0.f;
  for (int i = threadIdx.x; i < HW; i += 256) { float v = p[i]; s += v * v; }
  __shared__ float red[256];
  red[threadIdx.x] = s; __syncthreads();
  for (int off = 128; off > 0; off >>= 1) {
    if (threadIdx.x < off) red[threadIdx.x] += red[threadIdx.x + off];
    __syncthreads();
  }
  const float inv = 1.f / fmaxf(sqrtf(red[0]), 1e-12f);
  for (int i = threadIdx.x; i < HW; i += 256) p[i] *= inv;
}

// ---------------------------------------------------------------------------
// 5) rfft along the HW axis: radix-2 DIT, 16384 complex points in 128KB LDS
//    (CDNA5 WGP has 320KB LDS -> 2 workgroups/WGP). Keeps f=0..8192.
// ---------------------------------------------------------------------------
__global__ void k_fft_rows(const float* __restrict__ qkvd, int chOff,
                           float2* __restrict__ dst) {
  extern __shared__ float2 lds[];        // 16384 * 8B = 131072B dynamic LDS
  const int row = blockIdx.x;            // 0..1023 -> (b, head*64+c)
  const int b = row >> 8, local = row & 255;
  const float* in = qkvd + ((size_t)b * 768 + chOff + local) * HW;
  for (int i = threadIdx.x; i < HW; i += 256) {
    unsigned rv = __brev((unsigned)i) >> 18;   // 14-bit bit-reverse
    lds[i] = make_float2(in[rv], 0.f);
  }
  __syncthreads();
  for (int s = 1; s <= 14; ++s) {
    const int hm = 1 << (s - 1);
    for (int j = threadIdx.x; j < (HW >> 1); j += 256) {
      const int grp = j >> (s - 1);
      const int pos = j & (hm - 1);
      const int i1 = (grp << s) + pos;
      const int i2 = i1 + hm;
      float sn, cn;
      __sincosf(-6.2831853071795864f * (float)pos / (float)(hm << 1), &sn, &cn);
      float2 u = lds[i1], t = lds[i2];
      float tr = t.x * cn - t.y * sn;
      float ti = t.x * sn + t.y * cn;
      lds[i1] = make_float2(u.x + tr, u.y + ti);
      lds[i2] = make_float2(u.x - tr, u.y - ti);
    }
    __syncthreads();
  }
  float2* d = dst + (size_t)row * 8193;
  for (int f = threadIdx.x; f <= 8192; f += 256) d[f] = lds[f];
}

// ---------------------------------------------------------------------------
// 6) length-64 DFT along the channel axis; emit f16 (re,im) for the WMMA GEMM
// ---------------------------------------------------------------------------
__global__ void k_cdft64(const float2* __restrict__ src, _Float16* __restrict__ dsth) {
  const int f = blockIdx.x * 256 + threadIdx.x;
  const int mu = blockIdx.y, h = blockIdx.z;
  __shared__ float cs[64], sn[64];
  if (threadIdx.x < 64) {
    float ang = 6.2831853071795864f * (float)threadIdx.x / 64.f;
    cs[threadIdx.x] = cosf(ang);
    sn[threadIdx.x] = sinf(ang);
  }
  __syncthreads();
  if (f > 8192) return;
  const float2* S = src + (size_t)h * 64 * 8193;
  float re = 0.f, im = 0.f;
  for (int c = 0; c < 64; ++c) {
    float2 v = S[(size_t)c * 8193 + f];
    int t = (mu * c) & 63;                      // e^{-2pi i mu c / 64}
    re +=  v.x * cs[t] + v.y * sn[t];
    im += -v.x * sn[t] + v.y * cs[t];
  }
  size_t o = (((size_t)h * 64 + mu) * 8193 + f) * 2;
  dsth[o]     = (_Float16)re;
  dsth[o + 1] = (_Float16)im;
}

// ---------------------------------------------------------------------------
// 7) attn_hat[mu,nu] = temp * sum_f Q[mu,f]*K[nu,f]  (complex, no conjugate)
//    4 real GEMMs (RR,II,RI,IR) via WMMA f16, K-dim 8193 chunked by 32.
// ---------------------------------------------------------------------------
__global__ void k_attn_fft_gemm(const _Float16* __restrict__ qh,
                                const _Float16* __restrict__ kh,
                                const float* __restrict__ temperature,
                                float2* __restrict__ ahat) {
  const int lane = threadIdx.x & 31;
  const int half = lane >> 4, r = lane & 15;
  const int n0 = blockIdx.x * 16, m0 = blockIdx.y * 16, h = blockIdx.z;
  const _Float16* Q = qh + (size_t)h * 64 * 8193 * 2;
  const _Float16* K = kh + (size_t)h * 64 * 8193 * 2;
  v8f aRR = {}, aII = {}, aRI = {}, aIR = {};
  for (int f0 = 0; f0 < 8193; f0 += 32) {
    v16h ar, ai, br, bi;
#pragma unroll
    for (int j = 0; j < 16; ++j) {
      int kk = f0 + wmma_kidx(j, half);
      bool ok = kk < 8193;
      size_t qi = ((size_t)(m0 + r) * 8193 + kk) * 2;
      size_t ki = ((size_t)(n0 + r) * 8193 + kk) * 2;
      ar[j] = ok ? Q[qi]     : (_Float16)0.f;
      ai[j] = ok ? Q[qi + 1] : (_Float16)0.f;
      br[j] = ok ? K[ki]     : (_Float16)0.f;
      bi[j] = ok ? K[ki + 1] : (_Float16)0.f;
    }
    aRR = __builtin_amdgcn_wmma_f32_16x16x32_f16(false, ar, false, br, (short)0, aRR, false, false);
    aII = __builtin_amdgcn_wmma_f32_16x16x32_f16(false, ai, false, bi, (short)0, aII, false, false);
    aRI = __builtin_amdgcn_wmma_f32_16x16x32_f16(false, ar, false, bi, (short)0, aRI, false, false);
    aIR = __builtin_amdgcn_wmma_f32_16x16x32_f16(false, ai, false, br, (short)0, aIR, false, false);
  }
  const float tp = temperature[h & 3];   // per-head temperature
  float2* A = ahat + (size_t)h * 64 * 64;
#pragma unroll
  for (int t = 0; t < 8; ++t) {
    int m = m0 + t + 8 * half, n = n0 + r;
    A[m * 64 + n] = make_float2((aRR[t] - aII[t]) * tp, (aRI[t] + aIR[t]) * tp);
  }
}

// ---------------------------------------------------------------------------
// 8) irfft2 with s=(64,64): crops last axis to nu<=32, Hermitian completion.
// ---------------------------------------------------------------------------
__global__ void k_irfft_attn(const float2* __restrict__ ahat, float* __restrict__ attn) {
  __shared__ float2 A[64 * 33];
  __shared__ float cs[64], sn[64];
  const int h = blockIdx.y;
  for (int i = threadIdx.x; i < 64 * 33; i += 256) {
    int mu = i / 33, nu = i % 33;
    A[i] = ahat[((size_t)h * 64 + mu) * 64 + nu];
  }
  if (threadIdx.x < 64) {
    float ang = 6.2831853071795864f * (float)threadIdx.x / 64.f;
    cs[threadIdx.x] = cosf(ang);
    sn[threadIdx.x] = sinf(ang);
  }
  __syncthreads();
  const int o = blockIdx.x * 256 + threadIdx.x;   // 0..4095
  const int c = o >> 6, d = o & 63;
  float acc = 0.f;
  for (int mu = 0; mu < 64; ++mu) {
    const float2* Arow = A + mu * 33;
    for (int nu = 0; nu <= 32; ++nu) {
      float w = (nu == 0 || nu == 32) ? 1.f : 2.f;
      int t = (mu * c + nu * d) & 63;
      float2 a = Arow[nu];
      acc += w * (a.x * cs[t] - a.y * sn[t]);
    }
  }
  attn[((size_t)h * 64 + c) * 64 + d] = acc * (1.f / 4096.f);
}

// ---------------------------------------------------------------------------
// 9) 4x top-k masked softmax collapsed into A_eff = sum_i w_i*softmax_i.
// ---------------------------------------------------------------------------
__global__ void k_topk_softmax(const float* __restrict__ attn,
                               const float* __restrict__ attn_w,
                               float* __restrict__ aeff) {
  const int row = blockIdx.x;     // h16*64 + c
  const int t = threadIdx.x;      // 0..63
  __shared__ float vals[64], byrank[64], red[64], denom[4];
  float v = attn[(size_t)row * 64 + t];
  vals[t] = v; red[t] = v;
  __syncthreads();
  int rank = 0;
  for (int j = 0; j < 64; ++j) {
    float u = vals[j];
    rank += (u > v) || (u == v && j < t);   // stable rank, ties by index
  }
  for (int off = 32; off > 0; off >>= 1) {
    if (t < off) red[t] = fmaxf(red[t], red[t + off]);
    __syncthreads();
  }
  const float m = red[0];
  const float e = expf(v - m);
  byrank[rank] = e;
  __syncthreads();
  if (t < 4) {
    const int kks[4] = {32, 42, 48, 51};    // c//2, 2c//3, 3c//4, 4c//5 (c=64)
    float s = 0.f;
    for (int j2 = 0; j2 < kks[t]; ++j2) s += byrank[j2];
    denom[t] = s;
  }
  __syncthreads();
  const int kks[4] = {32, 42, 48, 51};
  float o = 0.f;
#pragma unroll
  for (int i = 0; i < 4; ++i)
    if (rank < kks[i]) o += attn_w[i] * e / denom[i];
  aeff[(size_t)row * 64 + t] = o;
}

// ---------------------------------------------------------------------------
// 10) out_pre = A_eff @ v  (per head: [64x64]x[64x16384]), f16 result.
// ---------------------------------------------------------------------------
__global__ void k_av_gemm(const float* __restrict__ aeff, const float* __restrict__ qkvd,
                          _Float16* __restrict__ outp) {
  const int wave = threadIdx.x >> 5, lane = threadIdx.x & 31;
  const int half = lane >> 4, r = lane & 15;
  const int n0 = blockIdx.x * 128 + wave * 16;
  const int m0 = blockIdx.y * 16;
  const int h = blockIdx.z;
  const int b = h >> 2, hd = h & 3;
  const float* A = aeff + (size_t)h * 64 * 64;
  const float* V = qkvd + ((size_t)b * 768 + 512 + hd * 64) * HW;  // v slice
  v8f acc = {};
#pragma unroll
  for (int k0 = 0; k0 < 64; k0 += 32) {
    v16h av, bv;
#pragma unroll
    for (int j = 0; j < 16; ++j) {
      int kk = k0 + wmma_kidx(j, half);
      av[j] = (_Float16)A[(m0 + r) * 64 + kk];
      bv[j] = (_Float16)V[(size_t)kk * HW + n0 + r];
    }
    acc = __builtin_amdgcn_wmma_f32_16x16x32_f16(false, av, false, bv,
                                                 (short)0, acc, false, false);
  }
  _Float16* O = outp + ((size_t)b * 256 + hd * 64) * HW;
#pragma unroll
  for (int t = 0; t < 8; ++t)
    O[(size_t)(m0 + t + 8 * half) * HW + n0 + r] = (_Float16)acc[t];
}

// ---------------------------------------------------------------------------
// 11) Final 1x1 conv + bias: out[b,o,s] = sum_i Wo[o,i]*pre[b,i,s] + bo[o]
// ---------------------------------------------------------------------------
__global__ void k_out_gemm(const _Float16* __restrict__ Woh, const float* __restrict__ bo,
                           const _Float16* __restrict__ pre, float* __restrict__ out) {
  __shared__ _Float16 btile[8][2][32 * 16];
  const int wave = threadIdx.x >> 5, lane = threadIdx.x & 31;
  const int half = lane >> 4, r = lane & 15;
  const int n0 = blockIdx.x * 128 + wave * 16;
  const int m0 = blockIdx.y * 16;
  const int b  = blockIdx.z;
  const _Float16* X = pre + (size_t)b * NC * HW;
  v8f acc = gemm_k256_tile(Woh, X, &btile[wave][0][0], &btile[wave][1][0],
                           m0, n0, half, r);
  float* O = out + (size_t)b * NC * HW;
#pragma unroll
  for (int t = 0; t < 8; ++t) {
    int m = m0 + t + 8 * half;
    O[(size_t)m * HW + n0 + r] = acc[t] + bo[m];
  }
}

// ---------------------------------------------------------------------------
// Launch. Workspace layout (~422 MB), region A time-multiplexed:
//   [0,32M)      xn (f16)
//   [32M,224M)   region A: qkv f32 -> {fft tmp 68M | qfft_h 34M | kfft_h 34M |
//                out_pre f16 32M} after the depthwise conv consumes qkv
//   [224M,416M)  qkvd f32 (q,k l2-normalized in place; v read by av_gemm)
//   [416M,...)   ahat / attn / aeff / f16 weights (small)
// ---------------------------------------------------------------------------
extern "C" void kernel_launch(void* const* d_in, const int* in_sizes, int n_in,
                              void* d_out, int out_size, void* d_ws, size_t ws_size,
                              hipStream_t stream) {
  (void)in_sizes; (void)n_in; (void)out_size; (void)ws_size;
  const float* x      = (const float*)d_in[0];
  const float* gn_w   = (const float*)d_in[1];
  const float* gn_b   = (const float*)d_in[2];
  const float* w_qkv  = (const float*)d_in[3];
  const float* w_dw   = (const float*)d_in[4];
  const float* w_out  = (const float*)d_in[5];
  const float* b_out  = (const float*)d_in[6];
  const float* temp   = (const float*)d_in[7];
  const float* attn_w = (const float*)d_in[8];
  float* out = (float*)d_out;

  char* ws = (char*)d_ws;
  const size_t MB = 1ull << 20;
  _Float16* xn   = (_Float16*)ws;
  char* regA     = ws + 32 * MB;
  float*    qkv  = (float*)regA;
  float2*   tmpf = (float2*)regA;                 // alias: valid after dwconv
  _Float16* qfh  = (_Float16*)(regA + 68 * MB);
  _Float16* kfh  = (_Float16*)(regA + 102 * MB);
  _Float16* outp = (_Float16*)(regA + 136 * MB);
  float*    qkvd = (float*)(ws + 224 * MB);
  float2*   ahat = (float2*)(ws + 416 * MB);
  float*    attn = (float*)(ws + 417 * MB);
  float*    aeff = (float*)(ws + 418 * MB);
  _Float16* wqkv_h = (_Float16*)(ws + 419 * MB);  // 384 KB
  _Float16* wout_h = (_Float16*)(ws + 420 * MB);  // 128 KB

  k_cvt_f32_to_f16<<<(768 * 256 + 255) / 256, 256, 0, stream>>>(w_qkv, wqkv_h, 768 * 256);
  k_cvt_f32_to_f16<<<(256 * 256 + 255) / 256, 256, 0, stream>>>(w_out, wout_h, 256 * 256);
  k_groupnorm<<<128, 256, 0, stream>>>(x, gn_w, gn_b, xn);
  k_qkv_gemm<<<dim3(128, 48, 4), 256, 0, stream>>>(wqkv_h, xn, qkv);
  k_dwconv<<<dim3(64, 768, 4), 256, 0, stream>>>(qkv, w_dw, qkvd);
  k_l2norm<<<2048, 256, 0, stream>>>(qkvd);
  // q path: row rfft -> channel DFT (f16 output)
  k_fft_rows<<<1024, 256, 131072, stream>>>(qkvd, 0, tmpf);
  k_cdft64<<<dim3(33, 64, 16), 256, 0, stream>>>(tmpf, qfh);
  // k path
  k_fft_rows<<<1024, 256, 131072, stream>>>(qkvd, 256, tmpf);
  k_cdft64<<<dim3(33, 64, 16), 256, 0, stream>>>(tmpf, kfh);
  k_attn_fft_gemm<<<dim3(4, 4, 16), 32, 0, stream>>>(qfh, kfh, temp, ahat);
  k_irfft_attn<<<dim3(16, 16), 256, 0, stream>>>(ahat, attn);
  k_topk_softmax<<<1024, 64, 0, stream>>>(attn, attn_w, aeff);
  k_av_gemm<<<dim3(128, 4, 16), 256, 0, stream>>>(aeff, qkvd, outp);
  k_out_gemm<<<dim3(128, 16, 4), 256, 0, stream>>>(wout_h, b_out, outp, out);
}
